// XEyTransformerLayer_8203387536040
// MI455X (gfx1250) — matrix operand order repaired and checked by reading, not compiled, not run on hardware
//
#include <hip/hip_runtime.h>
#include <hip/hip_bf16.h>
#include <math.h>

// ---------------- constants ----------------
#define BSZ 2
#define SEQ 512
#define DXC 256
#define DEC 64
#define DYC 64
#define NHC 8

typedef _Float16 v16h __attribute__((ext_vector_type(16)));
typedef float    v8f  __attribute__((ext_vector_type(8)));

union F16Frag { unsigned u[8]; v16h v; };

__device__ __forceinline__ float gelu_f(float x) {
  return 0.5f * x * (1.f + erff(x * 0.70710678118654752f));
}
__device__ __forceinline__ unsigned pack2(float a, float b) {
  _Float16 ha = (_Float16)a, hb = (_Float16)b;
  unsigned short ua = __builtin_bit_cast(unsigned short, ha);
  unsigned short ub = __builtin_bit_cast(unsigned short, hb);
  return (unsigned)ua | ((unsigned)ub << 16);
}
__device__ __forceinline__ unsigned short f16b(float a) {
  _Float16 h = (_Float16)a;
  return __builtin_bit_cast(unsigned short, h);
}
__device__ __forceinline__ v8f zero8() {
  v8f z = {0.f,0.f,0.f,0.f,0.f,0.f,0.f,0.f};
  return z;
}

// ============================================================
// Generic batched WMMA GEMM:  C[z] = alpha * A[z] @ B[z] (+bias) (epilogue)
// A fp32 [M x K] (lda), B fp32 [K x N] (or [N x K] if bT), C fp32 [M x N].
// Dual-weight: for n >= nSplit read B2/bias2 (concatenate two linears).
// flags: 1=bias, 2=gelu, 4=row-mask (mask[m]), 8=edge-mask (mask over b,q,k)
// z = b*NH + h with per-b / per-h pointer strides.
// Staging is branchless: full tiles use float4 (global_load_b128); ragged
// edges use clamped addresses + select-to-zero (no exec-mask serialization).
// ============================================================
struct GemmP {
  const float* A; const float* B; const float* B2;
  const float* bias; const float* bias2; const float* mask;
  float* C;
  long long lda, ldb, ldb2, ldc;
  long long sAb, sAh, sBb, sBh, sCb, sCh;
  int M, N, K, NH, bT, nSplit, flags;
  float alpha;
};

__global__ __launch_bounds__(256) void gemm_wmma_kernel(GemmP p) {
  __shared__ unsigned As[64 * 17];
  __shared__ unsigned Bs[64 * 17];
  const int tid = threadIdx.x;
  const int z   = blockIdx.z;
  const int zb  = z / p.NH, zh = z % p.NH;
  const float* A = p.A + (long long)zb * p.sAb + (long long)zh * p.sAh;
  const float* B = p.B + (long long)zb * p.sBb + (long long)zh * p.sBh;
  float*       C = p.C + (long long)zb * p.sCb + (long long)zh * p.sCh;
  const int m0 = blockIdx.y * 64;
  const int n0 = blockIdx.x * 64;
  const int w = tid >> 5, lane = tid & 31;
  const int hf = lane >> 4, lr = lane & 15;
  const int wm = (w & 3) << 4;   // row offset of wave tile
  const int wn = (w >> 2) << 5;  // col offset of wave tile (2x 16-wide tiles)
  v8f c0 = zero8(), c1 = zero8();

  // staging geometry: 4 threads per row/col, 8 consecutive k each
  const int sr  = tid >> 2;        // 0..63
  const int skq = (tid & 3) << 3;  // 0,8,16,24

  const int  gmA = m0 + sr;
  const bool vmA = gmA < p.M;
  const float* Arow = A + (long long)(vmA ? gmA : 0) * p.lda;

  const int  gnB = n0 + sr;
  const bool vnB = gnB < p.N;
  const int  gnc = vnB ? gnB : (p.N - 1);
  const bool useB2 = (p.B2 != nullptr) && !p.bT && (gnc >= p.nSplit);
  const float* Bcol = p.bT ? (B + (long long)gnc * p.ldb)
                           : (useB2 ? (p.B2 + (gnc - p.nSplit)) : (B + gnc));
  const long long Bstep = useB2 ? p.ldb2 : p.ldb;

  for (int k0 = 0; k0 < p.K; k0 += 32) {
    // ---- stage A tile 64x32 -> f16 pairs ----
    float av[8];
    if (vmA && (k0 + skq + 8 <= p.K)) {
      const float4 x0 = *reinterpret_cast<const float4*>(Arow + k0 + skq);
      const float4 x1 = *reinterpret_cast<const float4*>(Arow + k0 + skq + 4);
      av[0]=x0.x; av[1]=x0.y; av[2]=x0.z; av[3]=x0.w;
      av[4]=x1.x; av[5]=x1.y; av[6]=x1.z; av[7]=x1.w;
    } else {
#pragma unroll
      for (int j = 0; j < 8; ++j) {
        int gk  = k0 + skq + j;
        int gkc = gk < p.K ? gk : (p.K - 1);
        float v = Arow[gkc];
        av[j] = (vmA && gk < p.K) ? v : 0.f;
      }
    }
#pragma unroll
    for (int j = 0; j < 4; ++j)
      As[sr * 17 + (skq >> 1) + j] = pack2(av[2*j], av[2*j+1]);

    // ---- stage B tile as [n][k] 64x32 -> f16 pairs ----
    float bv[8];
    if (p.bT) {
      if (vnB && (k0 + skq + 8 <= p.K)) {
        const float4 x0 = *reinterpret_cast<const float4*>(Bcol + k0 + skq);
        const float4 x1 = *reinterpret_cast<const float4*>(Bcol + k0 + skq + 4);
        bv[0]=x0.x; bv[1]=x0.y; bv[2]=x0.z; bv[3]=x0.w;
        bv[4]=x1.x; bv[5]=x1.y; bv[6]=x1.z; bv[7]=x1.w;
      } else {
#pragma unroll
        for (int j = 0; j < 8; ++j) {
          int gk  = k0 + skq + j;
          int gkc = gk < p.K ? gk : (p.K - 1);
          float v = Bcol[gkc];
          bv[j] = (vnB && gk < p.K) ? v : 0.f;
        }
      }
    } else {
#pragma unroll
      for (int j = 0; j < 8; ++j) {
        int gk  = k0 + skq + j;
        int gkc = gk < p.K ? gk : (p.K - 1);
        float v = Bcol[(long long)gkc * Bstep];
        bv[j] = (vnB && gk < p.K) ? v : 0.f;
      }
    }
#pragma unroll
    for (int j = 0; j < 4; ++j)
      Bs[sr * 17 + (skq >> 1) + j] = pack2(bv[2*j], bv[2*j+1]);

    if (k0 + 32 < p.K) __builtin_prefetch(Arow + k0 + 32 + skq, 0, 1);
    __syncthreads();

    // build fragments: element s of lane half hf holds k = 16*hf + s (contiguous order, same for A and B)
    F16Frag a, b0, b1;
#pragma unroll
    for (int v = 0; v < 8; ++v) {
      int pp = (hf << 3) + v; // K-pair index for A
      a.u[v]  = As[(wm + lr) * 17 + pp];
      b0.u[v] = Bs[(wn + lr) * 17 + (hf << 3) + v];
      b1.u[v] = Bs[(wn + 16 + lr) * 17 + (hf << 3) + v];
    }
    c0 = __builtin_amdgcn_wmma_f32_16x16x32_f16(false, a.v, false, b0.v, (short)0, c0, false, false);
    c1 = __builtin_amdgcn_wmma_f32_16x16x32_f16(false, a.v, false, b1.v, (short)0, c1, false, false);
    __syncthreads();
  }
  // epilogue; C layout: VGPR i -> M = i + 8*hf, N = lane&15
#pragma unroll
  for (int i = 0; i < 8; ++i) {
    int m = m0 + wm + i + (hf << 3);
    if (m >= p.M) continue;
#pragma unroll
    for (int t = 0; t < 2; ++t) {
      int n = n0 + wn + (t << 4) + lr;
      if (n >= p.N) continue;
      float v = ((t == 0) ? c0[i] : c1[i]) * p.alpha;
      if (p.flags & 1) {
        float bv2 = (p.B2 && !p.bT && n >= p.nSplit) ? p.bias2[n - p.nSplit] : p.bias[n];
        v += bv2;
      }
      if (p.flags & 2) v = gelu_f(v);
      if (p.flags & 4) v *= p.mask[m];
      if (p.flags & 8) {
        int bb = m >> 18, qq = (m >> 9) & 511, kk = m & 511;
        v *= p.mask[bb * 512 + qq] * p.mask[bb * 512 + kk];
      }
      C[(long long)m * p.ldc + n] = v;
    }
  }
}

// ============================================================
// Fused E-FFN + LayerNorm:  Eout = LN(Ein + W2@gelu(W1@Ein+b1)+b2)
// D=64, H=128; one block = 64 full rows; everything staged in LDS.
// ============================================================
__global__ __launch_bounds__(256) void effn_ln_kernel(
    const float* __restrict__ Ein, float* __restrict__ Eout,
    const float* __restrict__ w1, const float* __restrict__ b1,
    const float* __restrict__ w2, const float* __restrict__ b2,
    const float* __restrict__ gam, const float* __restrict__ bet) {
  __shared__ __align__(16) unsigned char smem_[25344 + 16640 + 2048 + 512];
  unsigned*       As  = (unsigned*)smem_;            // 64 x 33 pairs (A, dead after GEMM1)
  unsigned*       W1s = As + 64 * 33;                // 128 x 33 pairs (dead after GEMM1)
  unsigned*       W2s = (unsigned*)smem_;            // alias: 64 x 65 pairs
  unsigned short* Hs  = (unsigned short*)(smem_ + 25344); // 64 x 130 f16
  unsigned*       HsU = (unsigned*)(smem_ + 25344);  // 64 x 65 pairs
  float*          Cs  = (float*)(smem_ + 25344);     // alias Hs after GEMM2: 64 x 65
  float*          red = (float*)(smem_ + 25344 + 16640); // 2 x 64 x 4
  float*          mv  = (float*)(smem_ + 25344 + 16640 + 2048); // 64 x 2

  const int tid = threadIdx.x;
  const long long row0 = (long long)blockIdx.x * 64;
  const int w = tid >> 5, lane = tid & 31, hf = lane >> 4, lr = lane & 15;

  { // stage A 64x64 (vectorized float2 pairs)
    int r = tid >> 2; int kq = (tid & 3) << 4;
    const float2* arow2 = reinterpret_cast<const float2*>(Ein + (row0 + r) * 64);
#pragma unroll
    for (int j = 0; j < 8; ++j) {
      float2 t = arow2[(kq >> 1) + j];
      As[r*33 + (kq>>1) + j] = pack2(t.x, t.y);
    }
  }
  { // stage W1 (64x128) as [n][k]
    int n = tid >> 1; int kq = (tid & 1) << 5;
#pragma unroll
    for (int j = 0; j < 16; ++j) { int k = kq + 2*j; W1s[n*33 + (k>>1)] = pack2(w1[k*128+n], w1[(k+1)*128+n]); }
  }
  __syncthreads();
  { // GEMM1: 64x128, each wave 16 rows x 64 cols (4 tiles)
    v8f acc0 = zero8(), acc1 = zero8(), acc2 = zero8(), acc3 = zero8();
    const int wm = (w & 3) << 4;
    const int cb = (w >> 2) << 6;
#pragma unroll
    for (int ks = 0; ks < 2; ++ks) {
      int kph = ks << 4;
      F16Frag a;
#pragma unroll
      for (int v = 0; v < 8; ++v)
        a.u[v] = As[(wm+lr)*33 + kph + (hf<<3) + v];
      F16Frag b;
#pragma unroll
      for (int v = 0; v < 8; ++v) b.u[v] = W1s[(cb + 0  + lr)*33 + kph + (hf<<3) + v];
      acc0 = __builtin_amdgcn_wmma_f32_16x16x32_f16(false, a.v, false, b.v, (short)0, acc0, false, false);
#pragma unroll
      for (int v = 0; v < 8; ++v) b.u[v] = W1s[(cb + 16 + lr)*33 + kph + (hf<<3) + v];
      acc1 = __builtin_amdgcn_wmma_f32_16x16x32_f16(false, a.v, false, b.v, (short)0, acc1, false, false);
#pragma unroll
      for (int v = 0; v < 8; ++v) b.u[v] = W1s[(cb + 32 + lr)*33 + kph + (hf<<3) + v];
      acc2 = __builtin_amdgcn_wmma_f32_16x16x32_f16(false, a.v, false, b.v, (short)0, acc2, false, false);
#pragma unroll
      for (int v = 0; v < 8; ++v) b.u[v] = W1s[(cb + 48 + lr)*33 + kph + (hf<<3) + v];
      acc3 = __builtin_amdgcn_wmma_f32_16x16x32_f16(false, a.v, false, b.v, (short)0, acc3, false, false);
    }
    // hidden = gelu(acc + b1) -> f16 LDS
#pragma unroll
    for (int i = 0; i < 8; ++i) {
      int m = wm + i + (hf << 3);
      int n0c = cb + lr;
      Hs[m*130 + n0c +  0] = f16b(gelu_f(acc0[i] + b1[n0c +  0]));
      Hs[m*130 + n0c + 16] = f16b(gelu_f(acc1[i] + b1[n0c + 16]));
      Hs[m*130 + n0c + 32] = f16b(gelu_f(acc2[i] + b1[n0c + 32]));
      Hs[m*130 + n0c + 48] = f16b(gelu_f(acc3[i] + b1[n0c + 48]));
    }
  }
  __syncthreads();
  { // stage W2 (128x64) as [n][k]; overwrites As/W1s region
    int n = tid >> 2; int kq = (tid & 3) << 5;
#pragma unroll
    for (int j = 0; j < 16; ++j) { int k = kq + 2*j; W2s[n*65 + (k>>1)] = pack2(w2[k*64+n], w2[(k+1)*64+n]); }
  }
  __syncthreads();
  { // GEMM2: 64x64 = H(64x128) @ W2
    const int wm = (w & 3) << 4;
    const int wn = (w >> 2) << 5;
    v8f d0 = zero8(), d1 = zero8();
#pragma unroll
    for (int ks = 0; ks < 4; ++ks) {
      int kph = ks << 4;
      F16Frag a, bb0, bb1;
#pragma unroll
      for (int v = 0; v < 8; ++v) {
        a.u[v]   = HsU[(wm+lr)*65 + kph + (hf<<3) + v];
        bb0.u[v] = W2s[(wn+lr)*65 + kph + (hf<<3) + v];
        bb1.u[v] = W2s[(wn+16+lr)*65 + kph + (hf<<3) + v];
      }
      d0 = __builtin_amdgcn_wmma_f32_16x16x32_f16(false, a.v, false, bb0.v, (short)0, d0, false, false);
      d1 = __builtin_amdgcn_wmma_f32_16x16x32_f16(false, a.v, false, bb1.v, (short)0, d1, false, false);
    }
    __syncthreads();  // Hs reads done -> Cs alias safe
#pragma unroll
    for (int i = 0; i < 8; ++i) {
      int m = wm + i + (hf << 3);
      Cs[m*65 + wn + lr]      = d0[i] + b2[wn + lr];
      Cs[m*65 + wn + 16 + lr] = d1[i] + b2[wn + 16 + lr];
    }
  }
  __syncthreads();
  { // residual + LayerNorm over each 64-wide row
    int r = tid >> 2, part = tid & 3;
    const float* erow = Ein + (row0 + r) * 64;
    float s = 0.f, s2 = 0.f;
#pragma unroll
    for (int j = 0; j < 16; ++j) {
      int c = part * 16 + j;
      float x = erow[c] + Cs[r*65 + c];
      Cs[r*65 + c] = x;
      s += x; s2 += x * x;
    }
    red[(r<<2) + part] = s;
    red[256 + (r<<2) + part] = s2;
    __syncthreads();
    if (part == 0) {
      float ts  = red[r<<2] + red[(r<<2)+1] + red[(r<<2)+2] + red[(r<<2)+3];
      float ts2 = red[256+(r<<2)] + red[256+(r<<2)+1] + red[256+(r<<2)+2] + red[256+(r<<2)+3];
      float mean = ts * (1.f/64.f);
      float var  = ts2 * (1.f/64.f) - mean*mean;
      mv[r<<1] = mean;
      mv[(r<<1)+1] = rsqrtf(var + 1e-5f);
    }
    __syncthreads();
    float mean = mv[r<<1], rstd = mv[(r<<1)+1];
    float* orow = Eout + (row0 + r) * 64;
#pragma unroll
    for (int j = 0; j < 16; ++j) {
      int c = part * 16 + j;
      orow[c] = (Cs[r*65 + c] - mean) * rstd * gam[c] + bet[c];
    }
  }
}

// ---------------- LayerNorm(a+b) : blockDim.x == C, one row per block ----
__global__ __launch_bounds__(256) void ln_res_kernel(
    float* __restrict__ out, const float* __restrict__ a, const float* __restrict__ b,
    const float* __restrict__ gam, const float* __restrict__ bet, int C) {
  __shared__ float s1[256], s2[256];
  long long row = blockIdx.x;
  int tid = threadIdx.x;
  float x = a[row * C + tid] + b[row * C + tid];
  s1[tid] = x; s2[tid] = x * x;
  __syncthreads();
  for (int st = blockDim.x >> 1; st > 0; st >>= 1) {
    if (tid < st) { s1[tid] += s1[tid + st]; s2[tid] += s2[tid + st]; }
    __syncthreads();
  }
  float mean = s1[0] / C;
  float var  = s2[0] / C - mean * mean;
  float rstd = rsqrtf(var + 1e-5f);
  out[row * C + tid] = (x - mean) * rstd * gam[tid] + bet[tid];
}

// ---------------- softmax over k (512) with key mask ----------------
__global__ __launch_bounds__(256) void softmax_kernel(float* __restrict__ Y, const float* __restrict__ nm) {
  __shared__ float sm[256];
  long long row = blockIdx.x;          // (b*8+h)*512 + q
  int b = (int)(row >> 12);
  float* yr = Y + row * 512;
  int tid = threadIdx.x;
  bool m0v = nm[b * 512 + tid] > 0.f;
  bool m1v = nm[b * 512 + tid + 256] > 0.f;
  float v0 = m0v ? yr[tid]       : -3.4e38f;
  float v1 = m1v ? yr[tid + 256] : -3.4e38f;
  sm[tid] = fmaxf(v0, v1);
  __syncthreads();
  for (int st = 128; st > 0; st >>= 1) { if (tid < st) sm[tid] = fmaxf(sm[tid], sm[tid + st]); __syncthreads(); }
  float mx = sm[0];
  __syncthreads();
  float e0 = m0v ? __expf(v0 - mx) : 0.f;
  float e1 = m1v ? __expf(v1 - mx) : 0.f;
  sm[tid] = e0 + e1;
  __syncthreads();
  for (int st = 128; st > 0; st >>= 1) { if (tid < st) sm[tid] += sm[tid + st]; __syncthreads(); }
  float inv = 1.f / sm[0];
  yr[tid]       = e0 * inv;
  yr[tid + 256] = e1 * inv;
}

// ---------------- edge fuse: Y-modulation + newE8 ----------------
__global__ __launch_bounds__(256) void edge_fuse_kernel(
    float* __restrict__ Y, const float* __restrict__ E12, const float* __restrict__ G12,
    const float* __restrict__ GRN, const float* __restrict__ yeA, const float* __restrict__ yeM,
    const float* __restrict__ tc, float* __restrict__ nE8) {
  long long e = (long long)blockIdx.x * 256 + threadIdx.x;
  int b = (int)(e >> 18);
  int q = (int)((e >> 9) & 511);
  int k = (int)(e & 511);
  bool zm = (GRN[e * 64] == 0.f);
  float t = tc[b];
#pragma unroll
  for (int h = 0; h < 8; ++h) {
    long long yi = ((long long)(b * 8 + h) << 18) + ((long long)q << 9) + k;
    float y  = Y[yi];
    float ym = y * (E12[e * 16 + h] + 1.f) + E12[e * 16 + 8 + h];
    Y[yi] = ym;
    float ne = yeA[b * 8 + h] + (yeM[b * 8 + h] + 1.f) * ym;
    if (!zm) {
      float g1 = G12[e * 16 + h], g2 = G12[e * 16 + 8 + h];
      ne = t * ne * g1 + (1.f - t) * ne + t * g2;
    }
    nE8[e * 8 + h] = ne;
  }
}

// ---------------- t_cos from cosine schedule (computed analytically) ----
__global__ void tcos_kernel(const float* __restrict__ t_init, float* __restrict__ tc) {
  int b = threadIdx.x;
  if (b >= BSZ) return;
  int idx = (int)(t_init[b] * 1000.0f);
  const float s = 0.008f;
  const float hp = 1.57079632679489662f; // pi/2
  float f0 = hp * ((idx / 1001.0f) + s) / (1.f + s);
  float f1 = hp * (((idx + 1) / 1001.0f) + s) / (1.f + s);
  float c0 = cosf(f0), c1 = cosf(f1);
  float beta = 1.f - (c1 * c1) / (c0 * c0);
  beta = fminf(fmaxf(beta, 0.f), 0.9999f);
  tc[b] = 1.f - beta;
}

// ---------------- small dense linear (tiny M) ----------------
// flags: 1 = gelu, 2 = accumulate into out
__global__ void small_gemm_kernel(const float* __restrict__ A, const float* __restrict__ W,
                                  const float* __restrict__ bias, float* __restrict__ out,
                                  int M, int K, int N, int flags) {
  int i = blockIdx.x * blockDim.x + threadIdx.x;
  if (i >= M * N) return;
  int m = i / N, n = i % N;
  float acc = bias ? bias[n] : 0.f;
  for (int k = 0; k < K; ++k) acc += A[m * K + k] * W[k * N + n];
  if (flags & 1) acc = gelu_f(acc);
  if (flags & 2) acc += out[i];
  out[i] = acc;
}

// ---------------- newX affine: X = add[b] + (mul[b]+1)*X ----------------
__global__ void affine_x_kernel(float* __restrict__ X, const float* __restrict__ addv,
                                const float* __restrict__ mulv) {
  long long i = (long long)blockIdx.x * 256 + threadIdx.x;
  int b = (int)(i >> 17);
  int c = (int)(i & 255);
  X[i] = addv[b * 256 + c] + (mulv[b * 256 + c] + 1.f) * X[i];
}

// ---------------- X stats: mean|min|max|std(ddof=1) over N=512 ----------
__global__ void xstats_kernel(const float* __restrict__ X, float* __restrict__ out) {
  __shared__ float ss[128], ss2[128], smn[128], smx[128];
  int b = blockIdx.x >> 8, d = blockIdx.x & 255;
  int tid = threadIdx.x;
  float s = 0.f, s2 = 0.f, mn = 3.4e38f, mx = -3.4e38f;
  for (int i = tid; i < 512; i += 128) {
    float v = X[((long long)b * 512 + i) * 256 + d];
    s += v; s2 += v * v; mn = fminf(mn, v); mx = fmaxf(mx, v);
  }
  ss[tid] = s; ss2[tid] = s2; smn[tid] = mn; smx[tid] = mx;
  __syncthreads();
  for (int st = 64; st > 0; st >>= 1) {
    if (tid < st) { ss[tid]+=ss[tid+st]; ss2[tid]+=ss2[tid+st]; smn[tid]=fminf(smn[tid],smn[tid+st]); smx[tid]=fmaxf(smx[tid],smx[tid+st]); }
    __syncthreads();
  }
  if (tid == 0) {
    float stdv = sqrtf(fmaxf(0.f, (ss2[0] - ss[0]*ss[0]/512.f) / 511.f));
    out[b*1024 + d]       = ss[0] / 512.f;
    out[b*1024 + 256 + d] = smn[0];
    out[b*1024 + 512 + d] = smx[0];
    out[b*1024 + 768 + d] = stdv;
  }
}

// ---------------- E stats over N*N = 262144 ----------------
__global__ void estats_kernel(const float* __restrict__ E, float* __restrict__ out) {
  __shared__ float ss[256], ss2[256], smn[256], smx[256];
  int b = blockIdx.x >> 6, c = blockIdx.x & 63;
  int tid = threadIdx.x;
  float s = 0.f, s2 = 0.f, mn = 3.4e38f, mx = -3.4e38f;
  for (int i = tid; i < 262144; i += 256) {
    float v = E[((long long)b * 262144 + i) * 64 + c];
    s += v; s2 += v * v; mn = fminf(mn, v); mx = fmaxf(mx, v);
  }
  ss[tid] = s; ss2[tid] = s2; smn[tid] = mn; smx[tid] = mx;
  __syncthreads();
  for (int st = 128; st > 0; st >>= 1) {
    if (tid < st) { ss[tid]+=ss[tid+st]; ss2[tid]+=ss2[tid+st]; smn[tid]=fminf(smn[tid],smn[tid+st]); smx[tid]=fmaxf(smx[tid],smx[tid+st]); }
    __syncthreads();
  }
  if (tid == 0) {
    float n = 262144.f;
    float stdv = sqrtf(fmaxf(0.f, (ss2[0] - ss[0]*ss[0]/n) / (n - 1.f)));
    out[b*256 + c]       = ss[0] / n;
    out[b*256 + 64 + c]  = smn[0];
    out[b*256 + 128 + c] = smx[0];
    out[b*256 + 192 + c] = stdv;
  }
}

// ============================================================
// Host orchestration
// ============================================================
// input indices (jax pytree flatten: dict keys sorted; leaves {'b','w'} / {'b','g'})
enum {
  IN_X0 = 0, IN_X1, IN_E, IN_Y, IN_T, IN_MASK, IN_GRN,
  A_BGADD_B = 7, A_BGADD_W, A_BGMUL_B, A_BGMUL_W,
  A_EADD_B, A_EADD_W, A_EMUL_B, A_EMUL_W,
  A_EOUT_B, A_EOUT_W, A_EY_B, A_EY_W,
  A_K_B, A_K_W, A_Q_B, A_Q_W, A_V_B, A_V_W,
  A_XOUT_B, A_XOUT_W, A_XY_B, A_XY_W,
  A_YEADD_B, A_YEADD_W, A_YEMUL_B, A_YEMUL_W,
  A_YOUT1_B, A_YOUT1_W, A_YOUT2_B, A_YOUT2_W,
  A_YXADD_B, A_YXADD_W, A_YXMUL_B, A_YXMUL_W,
  A_YY_B, A_YY_W,
  T_LINE1_B = 43, T_LINE1_W, T_LINE2_B, T_LINE2_W,
  T_LINX1_B, T_LINX1_W, T_LINX2_B, T_LINX2_W,
  T_LINY1_B, T_LINY1_W, T_LINY2_B, T_LINY2_W,
  T_NE1_B, T_NE1_G, T_NE2_B, T_NE2_G,
  T_NX1_B, T_NX1_G, T_NX2_B, T_NX2_G,
  T_NY1_B, T_NY1_G, T_NY2_B, T_NY2_G
};
#define TL2OFF 24

static inline void gemm_launch(hipStream_t s,
    const float* A, long long lda, long long sAb, long long sAh,
    const float* B, long long ldb, long long sBb, long long sBh, int bT,
    const float* B2, long long ldb2, int nSplit,
    const float* bias, const float* bias2,
    float* C, long long ldc, long long sCb, long long sCh,
    int M, int N, int K, int NH, int Z, float alpha, int flags, const float* mask) {
  GemmP p;
  p.A = A; p.B = B; p.B2 = B2; p.bias = bias; p.bias2 = bias2; p.mask = mask; p.C = C;
  p.lda = lda; p.ldb = ldb; p.ldb2 = ldb2; p.ldc = ldc;
  p.sAb = sAb; p.sAh = sAh; p.sBb = sBb; p.sBh = sBh; p.sCb = sCb; p.sCh = sCh;
  p.M = M; p.N = N; p.K = K; p.NH = NH; p.bT = bT; p.nSplit = nSplit; p.flags = flags; p.alpha = alpha;
  dim3 grid((N + 63) / 64, (M + 63) / 64, Z);
  gemm_wmma_kernel<<<grid, dim3(256), 0, s>>>(p);
}

static inline void sgemm_launch(hipStream_t s, const float* A, const float* W, const float* bias,
                                float* out, int M, int K, int N, int flags) {
  int tot = M * N;
  small_gemm_kernel<<<(tot + 255) / 256, 256, 0, s>>>(A, W, bias, out, M, K, N, flags);
}

extern "C" void kernel_launch(void* const* d_in, const int* in_sizes, int n_in,
                              void* d_out, int out_size, void* d_ws, size_t ws_size,
                              hipStream_t stream) {
  (void)in_sizes; (void)n_in; (void)out_size; (void)ws_size;
  auto F = [&](int i) { return (const float*)d_in[i]; };
  float* ws  = (float*)d_ws;
  float* out = (float*)d_out;

  // workspace layout (floats)
  float* wsQ    = ws;                     // 262144
  float* wsK    = ws + 262144;            // 262144
  float* wsV    = ws + 524288;            // 262144
  float* wsNEWX = ws + 786432;            // 262144
  float* wsNX   = ws + 1048576;           // 262144
  float* wsXHID = ws + 1310720;           // 1048576
  float* wsXTMP = ws + 2359296;           // 262144
  float* wsY    = ws + 2621440;           // 4194304  (b,h,q,k)
  float* wsE12  = ws + 6815744;           // 8388608
  float* wsG12  = ws + 15204352;          // 8388608
  float* wsNE8  = ws + 23592960;          // 4194304
  float* wsNE   = ws + 27787264;          // 33554432
  float* small_ = ws + 61341696;
  float* wsTC  = small_ + 0;
  float* wsYEA = small_ + 8;
  float* wsYEM = small_ + 24;
  float* wsYXA = small_ + 64;
  float* wsYXM = small_ + 576;
  float* wsXST = small_ + 1088;   // (2,1024)
  float* wsEST = small_ + 3136;   // (2,256)
  float* wsYCUR = small_ + 3648;
  float* wsYH  = small_ + 3776;   // up to (2,256)
  float* wsYT  = small_ + 4288;
  float* wsYV1 = small_ + 4416;
  float* wsYV2 = small_ + 4544;
  float* wsNY  = small_ + 4672;

  float* outX0 = out;
  float* outX1 = out + 262144;
  float* outE  = out + 524288;
  float* outY  = out + 34078720;

  const float* nm = F(IN_MASK);

  // ---- scalars & y conditioning vectors ----
  tcos_kernel<<<1, 32, 0, stream>>>(F(IN_T), wsTC);
  sgemm_launch(stream, F(IN_Y), F(A_YEADD_W), F(A_YEADD_B), wsYEA, BSZ, 64, 8, 0);
  sgemm_launch(stream, F(IN_Y), F(A_YEMUL_W), F(A_YEMUL_B), wsYEM, BSZ, 64, 8, 0);
  sgemm_launch(stream, F(IN_Y), F(A_YXADD_W), F(A_YXADD_B), wsYXA, BSZ, 64, 256, 0);
  sgemm_launch(stream, F(IN_Y), F(A_YXMUL_W), F(A_YXMUL_B), wsYXM, BSZ, 64, 256, 0);

  // ---- projections (WMMA) ----
  gemm_launch(stream, F(IN_X0), 256, 0, 0, F(A_Q_W), 256, 0, 0, 0, nullptr, 0, 256,
              F(A_Q_B), nullptr, wsQ, 256, 0, 0, 1024, 256, 256, 1, 1, 1.f, 1 | 4, nm);
  gemm_launch(stream, F(IN_X1), 256, 0, 0, F(A_K_W), 256, 0, 0, 0, nullptr, 0, 256,
              F(A_K_B), nullptr, wsK, 256, 0, 0, 1024, 256, 256, 1, 1, 1.f, 1 | 4, nm);
  gemm_launch(stream, F(IN_X1), 256, 0, 0, F(A_V_W), 256, 0, 0, 0, nullptr, 0, 256,
              F(A_V_B), nullptr, wsV, 256, 0, 0, 1024, 256, 256, 1, 1, 1.f, 1 | 4, nm);

  // ---- edge linears fused [mul|add] -> N=16, edge-masked ----
  gemm_launch(stream, F(IN_E), 64, 0, 0, F(A_EMUL_W), 8, 0, 0, 0, F(A_EADD_W), 8, 8,
              F(A_EMUL_B), F(A_EADD_B), wsE12, 16, 0, 0, 524288, 16, 64, 1, 1, 1.f, 1 | 8, nm);
  gemm_launch(stream, F(IN_GRN), 64, 0, 0, F(A_BGMUL_W), 8, 0, 0, 0, F(A_BGADD_W), 8, 8,
              F(A_BGMUL_B), F(A_BGADD_B), wsG12, 16, 0, 0, 524288, 16, 64, 1, 1, 1.f, 1 | 8, nm);

  // ---- attention scores: z = b*8+h, S = Q @ K^T / sqrt(32) ----
  gemm_launch(stream, wsQ, 256, 131072, 32, wsK, 256, 131072, 32, 1, nullptr, 0, 512,
              nullptr, nullptr, wsY, 512, 2097152, 262144, 512, 512, 32, NHC, 16,
              0.176776695296636893f, 0, nullptr);

  // ---- edge modulation + newE8, then masked softmax ----
  edge_fuse_kernel<<<2048, 256, 0, stream>>>(wsY, wsE12, wsG12, F(IN_GRN), wsYEA, wsYEM, wsTC, wsNE8);
  softmax_kernel<<<8192, 256, 0, stream>>>(wsY, nm);

  // ---- wV = attn @ V ----
  gemm_launch(stream, wsY, 512, 2097152, 262144, wsV, 256, 131072, 32, 0, nullptr, 0, 32,
              nullptr, nullptr, wsNEWX, 256, 131072, 32, 512, 32, 512, NHC, 16, 1.f, 0, nullptr);
  affine_x_kernel<<<1024, 256, 0, stream>>>(wsNEWX, wsYXA, wsYXM);

  // ---- nX = (newX @ x_out + b) * mask ; nE = (newE8 @ e_out + b) * em ----
  gemm_launch(stream, wsNEWX, 256, 0, 0, F(A_XOUT_W), 256, 0, 0, 0, nullptr, 0, 256,
              F(A_XOUT_B), nullptr, wsNX, 256, 0, 0, 1024, 256, 256, 1, 1, 1.f, 1 | 4, nm);
  gemm_launch(stream, wsNE8, 8, 0, 0, F(A_EOUT_W), 64, 0, 0, 0, nullptr, 0, 64,
              F(A_EOUT_B), nullptr, wsNE, 64, 0, 0, 524288, 64, 8, 1, 1, 1.f, 1 | 8, nm);

  // ---- stats + new_y chain ----
  xstats_kernel<<<512, 128, 0, stream>>>(F(IN_X1), wsXST);
  estats_kernel<<<128, 256, 0, stream>>>(F(IN_E), wsEST);
  sgemm_launch(stream, F(IN_Y), F(A_YY_W), F(A_YY_B), wsYCUR, BSZ, 64, 64, 0);
  sgemm_launch(stream, wsXST, F(A_XY_W), F(A_XY_B), wsYCUR, BSZ, 1024, 64, 2);
  sgemm_launch(stream, wsEST, F(A_EY_W), F(A_EY_B), wsYCUR, BSZ, 256, 64, 2);
  sgemm_launch(stream, wsYCUR, F(A_YOUT1_W), F(A_YOUT1_B), wsYH, BSZ, 64, 64, 1);
  sgemm_launch(stream, wsYH, F(A_YOUT2_W), F(A_YOUT2_B), wsNY, BSZ, 64, 64, 0);

  // ================= transformer layer 1 (X0 path) =================
  ln_res_kernel<<<1024, 256, 0, stream>>>(outX0, F(IN_X0), wsNX, F(T_NX1_G), F(T_NX1_B), 256);
  gemm_launch(stream, outX0, 256, 0, 0, F(T_LINX1_W), 1024, 0, 0, 0, nullptr, 0, 1024,
              F(T_LINX1_B), nullptr, wsXHID, 1024, 0, 0, 1024, 1024, 256, 1, 1, 1.f, 1 | 2, nullptr);
  gemm_launch(stream, wsXHID, 1024, 0, 0, F(T_LINX2_W), 256, 0, 0, 0, nullptr, 0, 256,
              F(T_LINX2_B), nullptr, wsXTMP, 256, 0, 0, 1024, 256, 1024, 1, 1, 1.f, 1, nullptr);
  ln_res_kernel<<<1024, 256, 0, stream>>>(outX0, outX0, wsXTMP, F(T_NX2_G), F(T_NX2_B), 256);

  ln_res_kernel<<<524288, 64, 0, stream>>>(outE, F(IN_E), wsNE, F(T_NE1_G), F(T_NE1_B), 64);
  effn_ln_kernel<<<8192, 256, 0, stream>>>(outE, outE, F(T_LINE1_W), F(T_LINE1_B),
                                           F(T_LINE2_W), F(T_LINE2_B), F(T_NE2_G), F(T_NE2_B));

  ln_res_kernel<<<2, 64, 0, stream>>>(wsYV1, F(IN_Y), wsNY, F(T_NY1_G), F(T_NY1_B), 64);
  sgemm_launch(stream, wsYV1, F(T_LINY1_W), F(T_LINY1_B), wsYH, BSZ, 64, 256, 1);
  sgemm_launch(stream, wsYH, F(T_LINY2_W), F(T_LINY2_B), wsYT, BSZ, 256, 64, 0);
  ln_res_kernel<<<2, 64, 0, stream>>>(wsYV2, wsYV1, wsYT, F(T_NY2_G), F(T_NY2_B), 64);

  // ================= transformer layer 2 (X1 path) =================
  ln_res_kernel<<<1024, 256, 0, stream>>>(outX1, F(IN_X1), wsNX, F(T_NX1_G + TL2OFF), F(T_NX1_B + TL2OFF), 256);
  gemm_launch(stream, outX1, 256, 0, 0, F(T_LINX1_W + TL2OFF), 1024, 0, 0, 0, nullptr, 0, 1024,
              F(T_LINX1_B + TL2OFF), nullptr, wsXHID, 1024, 0, 0, 1024, 1024, 256, 1, 1, 1.f, 1 | 2, nullptr);
  gemm_launch(stream, wsXHID, 1024, 0, 0, F(T_LINX2_W + TL2OFF), 256, 0, 0, 0, nullptr, 0, 256,
              F(T_LINX2_B + TL2OFF), nullptr, wsXTMP, 256, 0, 0, 1024, 256, 1024, 1, 1, 1.f, 1, nullptr);
  ln_res_kernel<<<1024, 256, 0, stream>>>(outX1, outX1, wsXTMP, F(T_NX2_G + TL2OFF), F(T_NX2_B + TL2OFF), 256);

  ln_res_kernel<<<524288, 64, 0, stream>>>(outE, outE, wsNE, F(T_NE1_G + TL2OFF), F(T_NE1_B + TL2OFF), 64);
  effn_ln_kernel<<<8192, 256, 0, stream>>>(outE, outE, F(T_LINE1_W + TL2OFF), F(T_LINE1_B + TL2OFF),
                                           F(T_LINE2_W + TL2OFF), F(T_LINE2_B + TL2OFF),
                                           F(T_NE2_G + TL2OFF), F(T_NE2_B + TL2OFF));

  ln_res_kernel<<<2, 64, 0, stream>>>(wsYV1, wsYV2, wsNY, F(T_NY1_G + TL2OFF), F(T_NY1_B + TL2OFF), 64);
  sgemm_launch(stream, wsYV1, F(T_LINY1_W + TL2OFF), F(T_LINY1_B + TL2OFF), wsYH, BSZ, 64, 256, 1);
  sgemm_launch(stream, wsYH, F(T_LINY2_W + TL2OFF), F(T_LINY2_B + TL2OFF), wsYT, BSZ, 256, 64, 0);
  ln_res_kernel<<<2, 64, 0, stream>>>(outY, wsYV1, wsYT, F(T_NY2_G + TL2OFF), F(T_NY2_B + TL2OFF), 64);
}
